// MB_projection_71725953843846
// MI455X (gfx1250) — matrix operand order repaired and checked by reading, not compiled, not run on hardware
//
#include <hip/hip_runtime.h>

// FlyHash projection + winner-take-all for MI455X (gfx1250, wave32, WMMA).
//
// act = inp(4096x512) @ W^T (W binary, 6 nnz/row from proj_idx), then per-row
// keep values >= 32nd-largest. GEMM via v_wmma_f32_16x16x32_f16 with an f16
// hi/lo split of inp (W entries exact in f16) for ~f32 accuracy. A staged to
// LDS with GLOBAL_LOAD_ASYNC_TO_LDS_B128, then converted once per element to
// f16 hi/lo in LDS so the WMMA loop has no conversion VALU in its chain.

typedef __attribute__((ext_vector_type(16))) _Float16 v16h;
typedef __attribute__((ext_vector_type(8)))  _Float16 v8h;
typedef __attribute__((ext_vector_type(4)))  _Float16 v4h;
typedef __attribute__((ext_vector_type(8)))  float    v8f;
typedef __attribute__((__vector_size__(16))) int      v4i;   // builtin's b128 unit

#define AS1 __attribute__((address_space(1)))
#define AS3 __attribute__((address_space(3)))

#if defined(__has_builtin)
#if __has_builtin(__builtin_amdgcn_global_load_async_to_lds_b128)
#define HAVE_ASYNC_LDS 1
#endif
#endif

#define BATCH 4096
#define INF   512
#define OUTF  10240
#define MT    64      // batch rows per block
#define NT    64      // output cols per block
#define KC    64      // K chunk staged in LDS

// ---------------------------------------------------------------------------
// Kernel 1: act = inp @ W^T via WMMA. 8 waves/block (4 M x 2 N), each wave
// owns two 16x16 C tiles sharing the A fragment. 40 KB static LDS.
// ---------------------------------------------------------------------------
__global__ __launch_bounds__(256) void flyhash_gemm(const float* __restrict__ inp,
                                                    const int*   __restrict__ proj,
                                                    float*       __restrict__ act)
{
  __shared__ __align__(16) float    Araw[MT][KC];   // 16 KB raw f32 A chunk
  __shared__ __align__(32) _Float16 Ah[MT][KC];     //  8 KB f16 high part
  __shared__ __align__(32) _Float16 Al[MT][KC];     //  8 KB f16 low residual
  __shared__ __align__(32) _Float16 Bs[NT][KC];     //  8 KB W tile, [col][k]

  const int t    = threadIdx.x;
  const int lane = t & 31;
  const int wave = t >> 5;
  const int bm0  = blockIdx.y * MT;
  const int bn0  = blockIdx.x * NT;
  const int m0   = (wave & 3) * 16;    // wave's row offset in block tile
  const int n0   = (wave >> 2) * 32;   // wave's col offset (covers 32 cols)

  // Cache projection indices: thread t (< 64) owns output column bn0+t.
  int pidx[6];
  if (t < NT) {
    const int* p = proj + (size_t)(bn0 + t) * 6;
#pragma unroll
    for (int j = 0; j < 6; ++j) pidx[j] = p[j];
  }

  v8f c0 = {};  // C tile at (m0, n0)
  v8f c1 = {};  // C tile at (m0, n0+16)

  for (int kc = 0; kc < INF; kc += KC) {
    __syncthreads();  // previous iteration's LDS reads complete

    // Stage A chunk (64 rows x 64 k of f32 = 16 KB) into LDS.
    // 1024 x b128; 256 threads x 4. Async copy when available.
#pragma unroll
    for (int i = 0; i < (MT * KC) / (4 * 256); ++i) {
      const int j  = t + i * 256;
      const int r  = j >> 4;          // row within tile (16 float4 per row)
      const int cc = (j & 15) << 2;   // k within chunk
      const float* gp = inp + (size_t)(bm0 + r) * INF + (kc + cc);
      float*       lp = &Araw[r][cc];
#if HAVE_ASYNC_LDS
      __builtin_amdgcn_global_load_async_to_lds_b128((AS1 v4i*)gp, (AS3 v4i*)lp,
                                                     0, 0);
#else
      *(float4*)lp = *(const float4*)gp;
#endif
    }

    if (kc + KC < INF)
      __builtin_prefetch(inp + (size_t)bm0 * INF + (kc + KC), 0, 1);

    // Zero the B tile (8 KB) with 128-bit stores.
    {
      float4 z; z.x = z.y = z.z = z.w = 0.0f;
      float4* bz = (float4*)&Bs[0][0];
#pragma unroll
      for (int j = 0; j < 2; ++j) bz[t + j * 256] = z;
    }

#if HAVE_ASYNC_LDS
#if __has_builtin(__builtin_amdgcn_s_wait_asynccnt)
    __builtin_amdgcn_s_wait_asynccnt(0);
#else
    asm volatile("s_wait_asynccnt 0" ::: "memory");
#endif
#endif
    __syncthreads();  // A staged, B zeroed

    // Convert raw f32 -> f16 hi/lo, once per element (LDS -> LDS).
#pragma unroll
    for (int i = 0; i < (MT * KC) / (4 * 256); ++i) {
      const int j  = t + i * 256;
      const int r  = j >> 4;
      const int cc = (j & 15) << 2;
      const float4 a = *(const float4*)&Araw[r][cc];
      const _Float16 h0 = (_Float16)a.x, h1 = (_Float16)a.y,
                     h2 = (_Float16)a.z, h3 = (_Float16)a.w;
      const v4h hv = {h0, h1, h2, h3};
      const v4h lv = {(_Float16)(a.x - (float)h0), (_Float16)(a.y - (float)h1),
                      (_Float16)(a.z - (float)h2), (_Float16)(a.w - (float)h3)};
      *(v4h*)&Ah[r][cc] = hv;
      *(v4h*)&Al[r][cc] = lv;
    }

    // Scatter W entries (disjoint from Ah/Al): thread t serially owns row t
    // of Bs -> duplicate indices accumulate correctly with no races.
    if (t < NT) {
#pragma unroll
      for (int j = 0; j < 6; ++j) {
        const int k = pidx[j] - kc;
        if ((unsigned)k < (unsigned)KC)
          Bs[t][k] = Bs[t][k] + (_Float16)1.0f;
      }
    }
    __syncthreads();  // tiles ready

    // 2 k-steps of 32; 4 WMMAs each (hi/lo x two N tiles, A frag shared).
#pragma unroll
    for (int ks = 0; ks < KC; ks += 32) {
      // A fragment in this code's contiguous k order (B matches): lanes 0-15 M=lane, halfs 0-15 ->
      // K=kb..kb+15, where kb = 16*(lane>=16).
      const int am = m0 + (lane & 15);
      const int ak = ks + ((lane >> 4) << 4);
      const v8h ah0 = *(const v8h*)&Ah[am][ak];
      const v8h ah1 = *(const v8h*)&Ah[am][ak + 8];
      const v8h al0 = *(const v8h*)&Al[am][ak];
      const v8h al1 = *(const v8h*)&Al[am][ak + 8];
      const v16h afh = __builtin_shufflevector(ah0, ah1,
          0,1,2,3,4,5,6,7,8,9,10,11,12,13,14,15);
      const v16h afl = __builtin_shufflevector(al0, al1,
          0,1,2,3,4,5,6,7,8,9,10,11,12,13,14,15);
      // B fragment, same contiguous k order as A: lane N = lane&15, K = 16*(lane>=16)+h.
      const int bk = ks + ((lane >> 4) << 4);
      const v16h bf0 = *(const v16h*)&Bs[n0 + (lane & 15)][bk];
      const v16h bf1 = *(const v16h*)&Bs[n0 + 16 + (lane & 15)][bk];

      c0 = __builtin_amdgcn_wmma_f32_16x16x32_f16(false, afh, false, bf0,
                                                  (short)0, c0, false, false);
      c0 = __builtin_amdgcn_wmma_f32_16x16x32_f16(false, afl, false, bf0,
                                                  (short)0, c0, false, false);
      c1 = __builtin_amdgcn_wmma_f32_16x16x32_f16(false, afh, false, bf1,
                                                  (short)0, c1, false, false);
      c1 = __builtin_amdgcn_wmma_f32_16x16x32_f16(false, afl, false, bf1,
                                                  (short)0, c1, false, false);
    }
  }

  // C layout: element e -> M = e + 8*(lane>=16), N = lane&15.
  const int n  = bn0 + n0 + (lane & 15);
  const int mr = bm0 + m0 + ((lane >> 4) << 3);
#pragma unroll
  for (int e = 0; e < 8; ++e) {
    act[(size_t)(mr + e) * OUTF + n]      = c0[e];
    act[(size_t)(mr + e) * OUTF + n + 16] = c1[e];
  }
}

// ---------------------------------------------------------------------------
// Kernel 2: exact per-row 32nd-largest via 4-pass radix select (8-bit digits).
// act >= 0, so uint32 bit order == float order. One block per row. Bin
// selection uses a parallel suffix scan (no serial 256-iteration loop).
// ---------------------------------------------------------------------------
__global__ __launch_bounds__(256) void flyhash_kth(const float* __restrict__ act,
                                                   float* __restrict__ kth,
                                                   const int* __restrict__ khash)
{
  __shared__ unsigned hist[256];
  __shared__ unsigned sel_digit;
  __shared__ unsigned sel_rank;

  const int t = threadIdx.x;
  const float* arow = act + (size_t)blockIdx.x * OUTF;
  unsigned prefix = 0;
  unsigned rank = (unsigned)*khash;  // 32

  for (int p = 3; p >= 0; --p) {
    hist[t] = 0;
    __syncthreads();
    const int shift = p * 8;
    for (int i = t; i < OUTF / 4; i += 256) {
      const float4 v = ((const float4*)arow)[i];
      const unsigned u[4] = {__float_as_uint(v.x), __float_as_uint(v.y),
                             __float_as_uint(v.z), __float_as_uint(v.w)};
#pragma unroll
      for (int j = 0; j < 4; ++j) {
        const bool match = (p == 3) || ((u[j] >> (shift + 8)) == prefix);
        if (match) atomicAdd(&hist[(u[j] >> shift) & 255u], 1u);
      }
    }
    __syncthreads();

    // Inclusive suffix sum over 256 bins (Hillis-Steele, 8 steps).
#pragma unroll
    for (int off = 1; off < 256; off <<= 1) {
      const unsigned mine = hist[t];
      const unsigned add  = (t + off < 256) ? hist[t + off] : 0u;
      __syncthreads();
      hist[t] = mine + add;
      __syncthreads();
    }
    // hist[b] = #elements with digit >= b (among prefix matches).
    const unsigned S  = hist[t];
    const unsigned Sn = (t < 255) ? hist[t + 1] : 0u;
    if (Sn < rank && rank <= S) {   // unique crossing thread
      sel_digit = (unsigned)t;
      sel_rank  = rank - Sn;
    }
    __syncthreads();
    prefix = (prefix << 8) | sel_digit;
    rank   = sel_rank;
    __syncthreads();
  }

  if (t == 0) kth[blockIdx.x] = __uint_as_float(prefix);
}

// ---------------------------------------------------------------------------
// Kernel 3: winner-take-all threshold, in place, float4 vectorized.
// ---------------------------------------------------------------------------
__global__ __launch_bounds__(256) void flyhash_thresh(float* __restrict__ act,
                                                      const float* __restrict__ kth)
{
  const size_t i = (size_t)blockIdx.x * blockDim.x + threadIdx.x;  // vec4 index
  const size_t nvec = (size_t)BATCH * OUTF / 4;
  if (i >= nvec) return;
  const int row = (int)(i / (OUTF / 4));
  const float kv = kth[row];
  float4 v = ((const float4*)act)[i];
  v.x = (v.x >= kv) ? v.x : 0.0f;
  v.y = (v.y >= kv) ? v.y : 0.0f;
  v.z = (v.z >= kv) ? v.z : 0.0f;
  v.w = (v.w >= kv) ? v.w : 0.0f;
  ((float4*)act)[i] = v;
}

// ---------------------------------------------------------------------------
extern "C" void kernel_launch(void* const* d_in, const int* in_sizes, int n_in,
                              void* d_out, int out_size, void* d_ws, size_t ws_size,
                              hipStream_t stream)
{
  const float* inp   = (const float*)d_in[0];
  const int*   proj  = (const int*)d_in[1];
  const int*   khash = (const int*)d_in[2];
  float*       out   = (float*)d_out;
  float*       kth   = (float*)d_ws;   // 4096 floats of scratch

  (void)in_sizes; (void)n_in; (void)out_size; (void)ws_size;

  dim3 g1(OUTF / NT, BATCH / MT);                 // 160 x 64 blocks
  flyhash_gemm<<<g1, 256, 0, stream>>>(inp, proj, out);
  flyhash_kth<<<BATCH, 256, 0, stream>>>(out, kth, khash);
  const size_t nvec = (size_t)BATCH * OUTF / 4;
  flyhash_thresh<<<(unsigned)((nvec + 255) / 256), 256, 0, stream>>>(out, kth);
}
